// STA_CNN_LSTM_987842478788
// MI455X (gfx1250) — matrix operand order repaired and checked by reading, not compiled, not run on hardware
//
#include <hip/hip_runtime.h>
#include <hip/hip_bf16.h>

typedef __attribute__((ext_vector_type(16))) _Float16 v16h;
typedef __attribute__((ext_vector_type(8)))  _Float16 v8h;
typedef __attribute__((ext_vector_type(8)))  float    v8f;

// ---------------------------------------------------------------------------
// helpers
// ---------------------------------------------------------------------------
__device__ __forceinline__ float sigm(float x) {
    return __builtin_amdgcn_rcpf(1.0f + __expf(-x));
}
__device__ __forceinline__ float tanh_fast(float x) {
    float ax = fabsf(x);
    float e  = __expf(-2.0f * ax);
    float r  = (1.0f - e) * __builtin_amdgcn_rcpf(1.0f + e);
    return copysignf(r, x);
}

__device__ __forceinline__ v8f wmma16(v16h a, v16h b, v8f c) {
    // D = A(16x32 f16) * B(32x16 f16) + C(16x16 f32)
    return __builtin_amdgcn_wmma_f32_16x16x32_f16(
        /*neg_a=*/false, a, /*neg_b=*/false, b,
        /*c_mod=*/(short)0, c, /*reuse_a=*/false, /*reuse_b=*/false);
}

// A fragment: 16x32 f16. lane L: M = m0 + (L&15); halves[0..7]=K kbase+8*(L>>4)..+7,
// halves[8..15]=K kbase+16+8*(L>>4)..+7  (per CDNA5 ISA 16-bit A layout)
__device__ __forceinline__ v16h load_afrag(const _Float16* base, int S, int m0,
                                           int kbase, int lane) {
    const int m  = m0 + (lane & 15);
    const int hi = lane >> 4;
    const _Float16* p = base + m * S + kbase + 8 * hi;
    v8h lo = *(const v8h*)p;
    v8h hh = *(const v8h*)(p + 16);
    return __builtin_shufflevector(lo, hh, 0, 1, 2, 3, 4, 5, 6, 7,
                                   8, 9, 10, 11, 12, 13, 14, 15);
}

// B fragment: 32x16 f16, weights stored [N][K] in LDS. lane L: N = n0+(L&15),
// halves[0..7]=K kbase+8*(L>>4)..+7, halves[8..15]=K kbase+16+8*(L>>4)..+7 (matches A).
__device__ __forceinline__ v16h load_bfrag(const _Float16* base, int S, int n0,
                                           int kbase, int lane) {
    const int n  = n0 + (lane & 15);
    const int hi = lane >> 4;
    const _Float16* p = base + n * S + kbase + 8 * hi;
    v8h lo = *(const v8h*)p;
    v8h hh = *(const v8h*)(p + 16);
    return __builtin_shufflevector(lo, hh, 0, 1, 2, 3, 4, 5, 6, 7,
                                   8, 9, 10, 11, 12, 13, 14, 15);
}

// C/D fragment -> f16 row-major LDS tile. lane L: N=n0+(L&15), rows m0+r+8*(L>>4)
__device__ __forceinline__ void store_frag_f16(_Float16* base, int S, int m0,
                                               int n0, int lane, v8f f) {
    const int n  = n0 + (lane & 15);
    const int hi = lane >> 4;
    _Float16* p = base + (m0 + 8 * hi) * S + n;
#pragma unroll
    for (int r = 0; r < 8; ++r) p[r * S] = (_Float16)f[r];
}

// ---------------------------------------------------------------------------
// Kernel 1: CNN + CBAM attention + fc1  (one wave32 per batch element)
// ---------------------------------------------------------------------------
struct CnnArgs {
    const float *x1, *w1, *cb1, *w2, *cb2, *wx, *cbx;
    const float *g1, *b1, *m1, *v1, *g2, *b2, *m2, *v2;
    const float *caf1, *caf2, *saw, *fc1w, *fc1b;
    float* out;  // (B,12)
};

__global__ __launch_bounds__(128) void sta_cnn_kernel(CnnArgs A) {
    __shared__ float a1s[4][32][9];
    __shared__ float Xs[4][64][9];
    __shared__ float cam[4][64], cax[4][64];
    __shared__ float hid[4][2][8];
    __shared__ float sab[4][2][9];
    __shared__ float ssb[4][9];

    const int tid = threadIdx.x;
    const int wave = tid >> 5, lane = tid & 31;
    const int b = blockIdx.x * 4 + wave;

    float xv[12];
    {
        const float* xin = A.x1 + (size_t)b * 12;
#pragma unroll
        for (int i = 0; i < 12; ++i) xv[i] = xin[i];  // [c*4 + iy*2 + ix]
    }
    // conv1 (2x2, pad 1 -> 3x3) + bn1 + relu; channel == lane
    {
        const float s1 = A.g1[lane] * rsqrtf(A.v1[lane] + 1e-5f);
        const float o1 = A.b1[lane] - A.m1[lane] * s1;
        const float* w = A.w1 + lane * 12;
        for (int y = 0; y < 3; ++y)
            for (int x = 0; x < 3; ++x) {
                float acc = A.cb1[lane];
#pragma unroll
                for (int c = 0; c < 3; ++c)
#pragma unroll
                    for (int ky = 0; ky < 2; ++ky)
#pragma unroll
                        for (int kx = 0; kx < 2; ++kx) {
                            const int iy = y + ky - 1, ix = x + kx - 1;
                            if (iy >= 0 && iy < 2 && ix >= 0 && ix < 2)
                                acc += xv[c * 4 + iy * 2 + ix] * w[c * 4 + ky * 2 + kx];
                        }
                a1s[wave][lane][y * 3 + x] = fmaxf(acc * s1 + o1, 0.f);
            }
    }
    __syncthreads();

    // conv2 (2x2, pad 0 -> 2x2) + bn2 + relu + maxpool(2,pad1 -> 3x3) + channel stats
    float P[2][9];
    for (int oo = 0; oo < 2; ++oo) {
        const int o = lane + 32 * oo;
        const float s2 = A.g2[o] * rsqrtf(A.v2[o] + 1e-5f);
        const float o2 = A.b2[o] - A.m2[o] * s2;
        float acc0 = A.cb2[o], acc1 = acc0, acc2 = acc0, acc3 = acc0;
        const float* w = A.w2 + (size_t)o * 128;
        for (int c = 0; c < 32; ++c) {
            const float w00 = w[c * 4 + 0], w01 = w[c * 4 + 1];
            const float w10 = w[c * 4 + 2], w11 = w[c * 4 + 3];
            const float* aa = a1s[wave][c];
            acc0 += aa[0] * w00 + aa[1] * w01 + aa[3] * w10 + aa[4] * w11;
            acc1 += aa[1] * w00 + aa[2] * w01 + aa[4] * w10 + aa[5] * w11;
            acc2 += aa[3] * w00 + aa[4] * w01 + aa[6] * w10 + aa[7] * w11;
            acc3 += aa[4] * w00 + aa[5] * w01 + aa[7] * w10 + aa[8] * w11;
        }
        const float X0 = fmaxf(acc0 * s2 + o2, 0.f), X1 = fmaxf(acc1 * s2 + o2, 0.f);
        const float X2 = fmaxf(acc2 * s2 + o2, 0.f), X3 = fmaxf(acc3 * s2 + o2, 0.f);
        float* Pp = P[oo];
        Pp[0] = X0;             Pp[1] = fmaxf(X0, X1);  Pp[2] = X1;
        Pp[3] = fmaxf(X0, X2);  Pp[4] = fmaxf(fmaxf(X0, X1), fmaxf(X2, X3));
        Pp[5] = fmaxf(X1, X3);  Pp[6] = X2;  Pp[7] = fmaxf(X2, X3);  Pp[8] = X3;
        float sm = 0.f, mx = -1e30f;
#pragma unroll
        for (int p = 0; p < 9; ++p) { sm += Pp[p]; mx = fmaxf(mx, Pp[p]); }
        cam[wave][o] = sm * (1.f / 9.f);
        cax[wave][o] = mx;
    }
    __syncthreads();

    // channel attention MLP: shared hidden layer (linearity of fc2 lets us sum relus)
    if (lane < 16) {
        const int hix = lane & 7;
        const float* src = (lane < 8) ? cam[wave] : cax[wave];
        float s = 0.f;
        for (int c = 0; c < 64; ++c) s += src[c] * A.caf1[hix * 64 + c];
        hid[wave][lane >> 3][hix] = fmaxf(s, 0.f);
    }
    __syncthreads();
    for (int oo = 0; oo < 2; ++oo) {
        const int o = lane + 32 * oo;
        float v = 0.f;
#pragma unroll
        for (int h = 0; h < 8; ++h)
            v += (hid[wave][0][h] + hid[wave][1][h]) * A.caf2[o * 8 + h];
        const float sc = sigm(v);
#pragma unroll
        for (int p = 0; p < 9; ++p) Xs[wave][o][p] = P[oo][p] * sc;
    }
    __syncthreads();

    // spatial attention: ch-mean/max -> 3x3 conv (k3,pad1) -> softmax over 9
    if (lane < 9) {
        float sm = 0.f, mx = -1e30f;
        for (int c = 0; c < 64; ++c) {
            const float t = Xs[wave][c][lane];
            sm += t; mx = fmaxf(mx, t);
        }
        sab[wave][0][lane] = sm * (1.f / 64.f);
        sab[wave][1][lane] = mx;
    }
    __syncthreads();
    if (lane < 9) {
        const int y = lane / 3, x = lane - y * 3;
        float acc = 0.f;
#pragma unroll
        for (int c2 = 0; c2 < 2; ++c2)
#pragma unroll
            for (int ky = 0; ky < 3; ++ky)
#pragma unroll
                for (int kx = 0; kx < 3; ++kx) {
                    const int iy = y + ky - 1, ix = x + kx - 1;
                    if (iy >= 0 && iy < 3 && ix >= 0 && ix < 3)
                        acc += sab[wave][c2][iy * 3 + ix] * A.saw[c2 * 9 + ky * 3 + kx];
                }
        ssb[wave][lane] = acc;
    }
    __syncthreads();
    float mxs = -1e30f;
#pragma unroll
    for (int p = 0; p < 9; ++p) mxs = fmaxf(mxs, ssb[wave][p]);
    float ssn[9], den = 0.f;
#pragma unroll
    for (int p = 0; p < 9; ++p) { ssn[p] = __expf(ssb[wave][p] - mxs); den += ssn[p]; }
    const float invd = __builtin_amdgcn_rcpf(den);

    // X*ss + sigmoid(convx(x1)) -> relu -> fc1 partials -> wave reduce
    float part[12];
#pragma unroll
    for (int d = 0; d < 12; ++d) part[d] = 0.f;
    for (int oo = 0; oo < 2; ++oo) {
        const int o = lane + 32 * oo;
        const float* wx = A.wx + o * 12;
        float xr[9];
        for (int y = 0; y < 3; ++y)
            for (int x = 0; x < 3; ++x) {
                const int p = y * 3 + x;
                float cx = A.cbx[o];
#pragma unroll
                for (int c = 0; c < 3; ++c)
#pragma unroll
                    for (int ky = 0; ky < 2; ++ky)
#pragma unroll
                        for (int kx = 0; kx < 2; ++kx) {
                            const int iy = y + ky - 1, ix = x + kx - 1;
                            if (iy >= 0 && iy < 2 && ix >= 0 && ix < 2)
                                cx += xv[c * 4 + iy * 2 + ix] * wx[c * 4 + ky * 2 + kx];
                        }
                xr[p] = fmaxf(Xs[wave][o][p] * ssn[p] * invd + sigm(cx), 0.f);
            }
        for (int d = 0; d < 12; ++d) {
            float s = 0.f;
#pragma unroll
            for (int p = 0; p < 9; ++p) s += xr[p] * A.fc1w[d * 576 + o * 9 + p];
            part[d] += s;
        }
    }
#pragma unroll
    for (int off = 16; off > 0; off >>= 1)
#pragma unroll
        for (int d = 0; d < 12; ++d) part[d] += __shfl_xor(part[d], off);
    if (lane == 0)
#pragma unroll
        for (int d = 0; d < 12; ++d)
            A.out[(size_t)b * 12 + d] = fmaxf(part[d] + A.fc1b[d], 0.f);
}

// ---------------------------------------------------------------------------
// Kernel 2: fused LSTM1 (input attention) + LSTM2 + temporal attention + heads
//   128 threads = 4 waves; 64 batch rows per block; everything LDS-resident.
//   All GEMMs via v_wmma_f32_16x16x32_f16 (f16 operands, f32 accumulate).
//   LSTM1 bias is folded into the GEMM via a constant-1 column (HX[:,67]).
// ---------------------------------------------------------------------------
struct LstmArgs {
    const float* x2;
    const float *Wih, *Whh, *bih, *bhh;
    const float* afc1w;
    const float *afc2w0, *afc2b0, *afc2w1, *afc2b1, *afc2w2, *afc2b2;
    const float *afc3w0, *afc3b0, *afc3w1, *afc3b1, *afc3w2, *afc3b2;
    const float *Wih2, *Whh2, *bih2, *bhh2;
    const float *f11w, *f11b, *f22w, *f22b;
    float* out;  // (B,12)
};

// LDS byte budget (dynamic)
static constexpr size_t LSTM_SMEM =
    (size_t)2 * (256 * 96 + 256 * 64 + 64 * 128 + 16 * 64 + 64 * 128 + 64 * 64 +
                 13 * 64 * 64) +
    (size_t)4 * (256 + 256 + 64 + 16 + 1536 + 432 + 36 + 36 + 4 + 2304);

__global__ __launch_bounds__(128) void sta_lstm_kernel(LstmArgs A) {
    extern __shared__ char smem[];
    _Float16* W1   = (_Float16*)smem;       // [256][96]  Whh | Wih | bias | 0-pad
    _Float16* W2   = W1 + 256 * 96;         // [256][64]  Wih2+Whh2 folded
    _Float16* F11W = W2 + 256 * 64;         // [64][128]
    _Float16* F22W = F11W + 64 * 128;       // [16][64] (rows 12..15 zero)
    _Float16* HX   = F22W + 16 * 64;        // [64][128]  [h | xt | 1 | pad] staging
    _Float16* CST  = HX + 64 * 128;         // [64][64]   c staging / g2 staging
    _Float16* WN   = CST + 64 * 64;         // [13][64][64] slot0=zeros, 1..11=h_t, 12=hq
    float* FB    = (float*)(WN + 13 * 64 * 64);
    float* B1s   = FB;              // 256 (unused in hot loop; bias folded into W1)
    float* B2s   = B1s + 256;       // 256
    float* F11B  = B2s + 256;       // 64
    float* F22B  = F11B + 64;       // 16
    float* AFC1W = F22B + 16;       // 12*128
    float* AFC2W = AFC1W + 1536;    // 3*12*12
    float* AFC2B = AFC2W + 432;     // 3*12
    float* AFC3W = AFC2B + 36;      // 3*12
    float* AFC3B = AFC3W + 36;      // 4
    float* X2S   = AFC3B + 4;       // 64*36

    const int tid  = threadIdx.x;
    const int lane = tid & 31, wave = tid >> 5;
    const int m0   = wave * 16;
    const int row0 = blockIdx.x * 64;

    __builtin_prefetch(A.Whh + tid * 4, 0, 0);
    __builtin_prefetch(A.Wih2 + tid * 4, 0, 0);

    // ---- x2 batch tile: async DMA straight into LDS (ASYNCcnt-tracked) ----
    {
        const float* gsrc = A.x2 + (size_t)row0 * 36;
        for (int i = tid; i < 64 * 36; i += 128) {
            const unsigned lds = (unsigned)(size_t)(X2S + i);
            const unsigned long long ga = (unsigned long long)(size_t)(gsrc + i);
            asm volatile("global_load_async_to_lds_b32 %0, %1, off"
                         :: "v"(lds), "v"(ga) : "memory");
        }
    }

    // ------------------- weight preload (f32 -> f16 LDS) -------------------
    for (int n = tid; n < 256; n += 128) {
        const float* whh = A.Whh + n * 64;
        for (int k = 0; k < 64; ++k) W1[n * 96 + k] = (_Float16)whh[k];
        const float* wih = A.Wih + n * 3;
        W1[n * 96 + 64] = (_Float16)wih[0];
        W1[n * 96 + 65] = (_Float16)wih[1];
        W1[n * 96 + 66] = (_Float16)wih[2];
        W1[n * 96 + 67] = (_Float16)(A.bih[n] + A.bhh[n]);  // bias via 1-column
        for (int k = 68; k < 96; ++k) W1[n * 96 + k] = (_Float16)0.f;
        const float* wa = A.Wih2 + n * 64;
        const float* wb = A.Whh2 + n * 64;
        for (int k = 0; k < 64; ++k) W2[n * 64 + k] = (_Float16)(wa[k] + wb[k]);
        B2s[n] = A.bih2[n] + A.bhh2[n];
    }
    for (int i = tid; i < 64 * 128; i += 128) F11W[i] = (_Float16)A.f11w[i];
    for (int i = tid; i < 16 * 64; i += 128) {
        const int n = i >> 6, k = i & 63;
        F22W[i] = (n < 12) ? (_Float16)A.f22w[n * 64 + k] : (_Float16)0.f;
    }
    for (int i = tid; i < 64; i += 128) F11B[i] = A.f11b[i];
    for (int i = tid; i < 16; i += 128) F22B[i] = (i < 12) ? A.f22b[i] : 0.f;
    for (int i = tid; i < 1536; i += 128) AFC1W[i] = A.afc1w[i];
    {
        const float* a2w[3] = {A.afc2w0, A.afc2w1, A.afc2w2};
        const float* a2b[3] = {A.afc2b0, A.afc2b1, A.afc2b2};
        const float* a3w[3] = {A.afc3w0, A.afc3w1, A.afc3w2};
        const float* a3b[3] = {A.afc3b0, A.afc3b1, A.afc3b2};
        for (int i = tid; i < 3 * 144; i += 128) {
            const int j = i / 144;
            AFC2W[i] = a2w[j][i - j * 144];
        }
        for (int i = tid; i < 36; i += 128) {
            const int j = i / 12;
            AFC2B[i] = a2b[j][i - j * 12];
        }
        for (int i = tid; i < 36; i += 128) {
            const int j = i / 12;
            AFC3W[i] = a3w[j][i - j * 12];
        }
        if (tid < 3) AFC3B[tid] = a3b[tid][0];
    }
    for (int i = tid; i < 64 * 128; i += 128) HX[i] = (_Float16)0.f;
    for (int i = tid; i < 64 * 64; i += 128) WN[i] = (_Float16)0.f;  // Wn slot 0 = zeros
    asm volatile("s_wait_asynccnt 0x0" ::: "memory");  // x2 tile resident in LDS
    __syncthreads();
    // xt(0) = x2[:,0,:]; constant-1 column for bias injection
    for (int i = tid; i < 64 * 3; i += 128) {
        const int r = i / 3, j = i - r * 3;
        HX[r * 128 + 64 + j] = (_Float16)X2S[r * 36 + j];
    }
    for (int r = tid; r < 64; r += 128) HX[r * 128 + 67] = (_Float16)1.0f;
    __syncthreads();

    v8f cst[4];
    {
        v8f z = {};
        cst[0] = z; cst[1] = z; cst[2] = z; cst[3] = z;
    }

    // --------------------------- LSTM1: 12 steps ---------------------------
    for (int t = 0; t < 12; ++t) {
        const v16h a0 = load_afrag(HX, 128, m0, 0, lane);
        const v16h a1 = load_afrag(HX, 128, m0, 32, lane);
        const v16h a2 = load_afrag(HX, 128, m0, 64, lane);
#pragma unroll
        for (int g = 0; g < 4; ++g) {  // gate groups: tiles {g, g+4, g+8, g+12}
            v8f acc[4];
#pragma unroll
            for (int q = 0; q < 4; ++q) {
                const int nt = g + 4 * q;
                v8f a = {};  // bias arrives through HX[:,67] * W1[:,67]
                a = wmma16(a0, load_bfrag(W1, 96, nt * 16, 0, lane), a);
                a = wmma16(a1, load_bfrag(W1, 96, nt * 16, 32, lane), a);
                a = wmma16(a2, load_bfrag(W1, 96, nt * 16, 64, lane), a);
                acc[q] = a;
            }
            v8f c2 = {}, hh = {};
#pragma unroll
            for (int r = 0; r < 8; ++r) {
                const float iv = sigm(acc[0][r]);
                const float fv = sigm(acc[1][r]);
                const float gv = tanh_fast(acc[2][r]);
                const float ov = sigm(acc[3][r]);
                const float cv = fv * cst[g][r] + iv * gv;
                c2[r] = cv;
                hh[r] = ov * tanh_fast(cv);
            }
            cst[g] = c2;
            store_frag_f16(HX, 128, m0, g * 16, lane, hh);
            store_frag_f16(CST, 64, m0, g * 16, lane, c2);
        }
        __syncthreads();
        if (t < 11) {  // input attention -> xt(t+1); 2 threads per row
            const int row = tid >> 1, h2 = tid & 1;
            float w1v[6];
#pragma unroll
            for (int d6 = 0; d6 < 6; ++d6) {
                const int d = h2 * 6 + d6;
                float s = 0.f;
                for (int k = 0; k < 64; ++k) s += (float)HX[row * 128 + k] * AFC1W[d * 128 + k];
                for (int k = 0; k < 64; ++k) s += (float)CST[row * 64 + k] * AFC1W[d * 128 + 64 + k];
                w1v[d6] = s;
            }
            float ej[3];
#pragma unroll
            for (int j = 0; j < 3; ++j) {
                float part = 0.f;
#pragma unroll
                for (int d6 = 0; d6 < 6; ++d6) {
                    const int d = h2 * 6 + d6;
                    float s = w1v[d6] + AFC2B[j * 12 + d];
#pragma unroll
                    for (int tt = 0; tt < 12; ++tt)
                        s += X2S[row * 36 + tt * 3 + j] * AFC2W[j * 144 + d * 12 + tt];
                    part += sigm(s) * AFC3W[j * 12 + d];
                }
                part += __shfl_xor(part, 1);
                ej[j] = (part + AFC3B[j]) * ((j == 2) ? 0.8f : 1.0f);
            }
            const float m = fmaxf(ej[0], fmaxf(ej[1], ej[2]));
            const float e0 = __expf(ej[0] - m), e1 = __expf(ej[1] - m), e2 = __expf(ej[2] - m);
            const float inv = __builtin_amdgcn_rcpf(e0 + e1 + e2);
            if (h2 == 0) {
                HX[row * 128 + 64] = (_Float16)(X2S[row * 36 + (t + 1) * 3 + 0] * e0 * inv);
                HX[row * 128 + 65] = (_Float16)(X2S[row * 36 + (t + 1) * 3 + 1] * e1 * inv);
                HX[row * 128 + 66] = (_Float16)(X2S[row * 36 + (t + 1) * 3 + 2] * e2 * inv);
            }
        }
        __syncthreads();
    }

    // --------------------------- LSTM2: 12 steps ---------------------------
    for (int t = 0; t < 12; ++t) {
        const _Float16* src = (t == 0) ? HX : (WN + t * 64 * 64);
        const int S = (t == 0) ? 128 : 64;
        const v16h a0 = load_afrag(src, S, m0, 0, lane);
        const v16h a1 = load_afrag(src, S, m0, 32, lane);
        _Float16* dst = WN + ((t < 11) ? (t + 1) : 12) * 64 * 64;
#pragma unroll
        for (int g = 0; g < 4; ++g) {
            v8f acc[4];
#pragma unroll
            for (int q = 0; q < 4; ++q) {
                const int nt = g + 4 * q;
                const float bv = B2s[nt * 16 + (lane & 15)];
                v8f a = {bv, bv, bv, bv, bv, bv, bv, bv};
                a = wmma16(a0, load_bfrag(W2, 64, nt * 16, 0, lane), a);
                a = wmma16(a1, load_bfrag(W2, 64, nt * 16, 32, lane), a);
                acc[q] = a;
            }
            v8f c2 = {}, hh = {};
#pragma unroll
            for (int r = 0; r < 8; ++r) {
                const float iv = sigm(acc[0][r]);
                const float fv = sigm(acc[1][r]);
                const float gv = tanh_fast(acc[2][r]);
                const float ov = sigm(acc[3][r]);
                const float cv = fv * cst[g][r] + iv * gv;
                c2[r] = cv;
                hh[r] = ov * tanh_fast(cv);
            }
            cst[g] = c2;
            store_frag_f16(dst, 64, m0, g * 16, lane, hh);
        }
        __syncthreads();
    }

    // --------------- temporal attention: h_atten = [attn | hq] -------------
    if (tid < 64) {
        const int row = tid;
        const _Float16* hq = WN + 12 * 64 * 64 + row * 64;
        float sc12[12];
        float mx = -1e30f;
        for (int tt = 0; tt < 12; ++tt) {
            const _Float16* wv = WN + tt * 64 * 64 + row * 64;
            float s = 0.f;
            for (int k = 0; k < 64; ++k) s += (float)hq[k] * (float)wv[k];
            sc12[tt] = s;
            mx = fmaxf(mx, s);
        }
        float den = 0.f;
#pragma unroll
        for (int tt = 0; tt < 12; ++tt) { sc12[tt] = __expf(sc12[tt] - mx); den += sc12[tt]; }
        const float inv = __builtin_amdgcn_rcpf(den);
        for (int k = 0; k < 64; ++k) {
            float av = 0.f;
#pragma unroll
            for (int tt = 0; tt < 12; ++tt)
                av += sc12[tt] * (float)(WN + tt * 64 * 64)[row * 64 + k];
            HX[row * 128 + k]      = (_Float16)(av * inv);
            HX[row * 128 + 64 + k] = hq[k];
        }
    }
    __syncthreads();

    // ----------------------- f11 (128->64) + relu --------------------------
    {
        const v16h a0 = load_afrag(HX, 128, m0, 0, lane);
        const v16h a1 = load_afrag(HX, 128, m0, 32, lane);
        const v16h a2 = load_afrag(HX, 128, m0, 64, lane);
        const v16h a3 = load_afrag(HX, 128, m0, 96, lane);
#pragma unroll
        for (int nt = 0; nt < 4; ++nt) {
            const float bv = F11B[nt * 16 + (lane & 15)];
            v8f a = {bv, bv, bv, bv, bv, bv, bv, bv};
            a = wmma16(a0, load_bfrag(F11W, 128, nt * 16, 0, lane), a);
            a = wmma16(a1, load_bfrag(F11W, 128, nt * 16, 32, lane), a);
            a = wmma16(a2, load_bfrag(F11W, 128, nt * 16, 64, lane), a);
            a = wmma16(a3, load_bfrag(F11W, 128, nt * 16, 96, lane), a);
#pragma unroll
            for (int r = 0; r < 8; ++r) a[r] = fmaxf(a[r], 0.f);
            store_frag_f16(CST, 64, m0, nt * 16, lane, a);  // CST reused as g2
        }
    }
    __syncthreads();
    // ----------------------- f22 (64->12) + write out ----------------------
    {
        const v16h a0 = load_afrag(CST, 64, m0, 0, lane);
        const v16h a1 = load_afrag(CST, 64, m0, 32, lane);
        const float bv = F22B[lane & 15];
        v8f a = {bv, bv, bv, bv, bv, bv, bv, bv};
        a = wmma16(a0, load_bfrag(F22W, 64, 0, 0, lane), a);
        a = wmma16(a1, load_bfrag(F22W, 64, 0, 32, lane), a);
        const int n = lane & 15, hi = lane >> 4;
        if (n < 12) {
#pragma unroll
            for (int r = 0; r < 8; ++r) {
                const int row = row0 + m0 + 8 * hi + r;
                A.out[(size_t)row * 12 + n] = a[r];
            }
        }
    }
}

// ---------------------------------------------------------------------------
// Kernel 3: y = concat([Xcnn, out_lstm]) @ fc4.T + b
// ---------------------------------------------------------------------------
__global__ __launch_bounds__(256) void sta_fc4_kernel(const float* Xc, const float* Ol,
                                                      const float* w, const float* bias,
                                                      float* out, int B) {
    const int i = blockIdx.x * blockDim.x + threadIdx.x;
    if (i >= B) return;
    float v[24];
#pragma unroll
    for (int d = 0; d < 12; ++d) v[d] = Xc[(size_t)i * 12 + d];
#pragma unroll
    for (int d = 0; d < 12; ++d) v[12 + d] = Ol[(size_t)i * 12 + d];
#pragma unroll
    for (int o = 0; o < 4; ++o) {
        float s = bias[o];
#pragma unroll
        for (int k = 0; k < 24; ++k) s += v[k] * w[o * 24 + k];
        out[(size_t)i * 4 + o] = s;
    }
}

// ---------------------------------------------------------------------------
extern "C" void kernel_launch(void* const* d_in, const int* in_sizes, int n_in,
                              void* d_out, int out_size, void* d_ws, size_t ws_size,
                              hipStream_t stream) {
    (void)n_in; (void)out_size; (void)ws_size;
    const float* x1      = (const float*)d_in[0];
    const float* x2      = (const float*)d_in[1];
    const float* conv1_w = (const float*)d_in[2];
    const float* conv1_b = (const float*)d_in[3];
    const float* conv2_w = (const float*)d_in[4];
    const float* conv2_b = (const float*)d_in[5];
    const float* convx_w = (const float*)d_in[6];
    const float* convx_b = (const float*)d_in[7];
    const float* bn1_g   = (const float*)d_in[8];
    const float* bn1_b   = (const float*)d_in[9];
    const float* bn1_m   = (const float*)d_in[10];
    const float* bn1_v   = (const float*)d_in[11];
    const float* bn2_g   = (const float*)d_in[12];
    const float* bn2_b   = (const float*)d_in[13];
    const float* bn2_m   = (const float*)d_in[14];
    const float* bn2_v   = (const float*)d_in[15];
    const float* ca_fc1  = (const float*)d_in[16];
    const float* ca_fc2  = (const float*)d_in[17];
    const float* sa_w    = (const float*)d_in[18];
    const float* fc1_w   = (const float*)d_in[19];
    const float* fc1_b   = (const float*)d_in[20];
    const float* Wih     = (const float*)d_in[21];
    const float* Whh     = (const float*)d_in[22];
    const float* bih     = (const float*)d_in[23];
    const float* bhh     = (const float*)d_in[24];
    const float* afc1_w  = (const float*)d_in[25];
    const float* afc20_w = (const float*)d_in[26];
    const float* afc20_b = (const float*)d_in[27];
    const float* afc21_w = (const float*)d_in[28];
    const float* afc21_b = (const float*)d_in[29];
    const float* afc22_w = (const float*)d_in[30];
    const float* afc22_b = (const float*)d_in[31];
    const float* afc30_w = (const float*)d_in[32];
    const float* afc30_b = (const float*)d_in[33];
    const float* afc31_w = (const float*)d_in[34];
    const float* afc31_b = (const float*)d_in[35];
    const float* afc32_w = (const float*)d_in[36];
    const float* afc32_b = (const float*)d_in[37];
    const float* Wih2    = (const float*)d_in[38];
    const float* Whh2    = (const float*)d_in[39];
    const float* bih2    = (const float*)d_in[40];
    const float* bhh2    = (const float*)d_in[41];
    const float* f11_w   = (const float*)d_in[42];
    const float* f11_b   = (const float*)d_in[43];
    const float* f22_w   = (const float*)d_in[44];
    const float* f22_b   = (const float*)d_in[45];
    const float* fc4_w   = (const float*)d_in[46];
    const float* fc4_b   = (const float*)d_in[47];

    const int B = in_sizes[0] / 12;  // x1 is (B,3,2,2)
    float* Xc = (float*)d_ws;            // (B,12) CNN features
    float* Ol = Xc + (size_t)B * 12;     // (B,12) LSTM head output

    CnnArgs ca = {x1, conv1_w, conv1_b, conv2_w, conv2_b, convx_w, convx_b,
                  bn1_g, bn1_b, bn1_m, bn1_v, bn2_g, bn2_b, bn2_m, bn2_v,
                  ca_fc1, ca_fc2, sa_w, fc1_w, fc1_b, Xc};
    sta_cnn_kernel<<<B / 4, 128, 0, stream>>>(ca);

    LstmArgs la = {x2, Wih, Whh, bih, bhh, afc1_w,
                   afc20_w, afc20_b, afc21_w, afc21_b, afc22_w, afc22_b,
                   afc30_w, afc30_b, afc31_w, afc31_b, afc32_w, afc32_b,
                   Wih2, Whh2, bih2, bhh2, f11_w, f11_b, f22_w, f22_b, Ol};
    hipFuncSetAttribute(reinterpret_cast<const void*>(sta_lstm_kernel),
                        hipFuncAttributeMaxDynamicSharedMemorySize, (int)LSTM_SMEM);
    sta_lstm_kernel<<<B / 64, 128, LSTM_SMEM, stream>>>(la);

    sta_fc4_kernel<<<(B + 255) / 256, 256, 0, stream>>>(Xc, Ol, fc4_w, fc4_b,
                                                        (float*)d_out, B);
}
